// CrossBranchAttentionWithSA_3874060501747
// MI455X (gfx1250) — matrix operand order repaired and checked by reading, not compiled, not run on hardware
//
#include <hip/hip_runtime.h>
#include <hip/hip_bf16.h>
#include <stdint.h>

// ---------------- types ----------------
typedef __attribute__((ext_vector_type(16))) __bf16 v16bf;
typedef __attribute__((ext_vector_type(8)))  float  v8f;
typedef __attribute__((ext_vector_type(4)))  unsigned int u32x4;

struct Pack32 { u32x4 a, b; };

static __device__ __forceinline__ v16bf load_v16(const unsigned short* p0,
                                                 const unsigned short* p1) {
  Pack32 s;
  s.a = *reinterpret_cast<const u32x4*>(p0);
  s.b = *reinterpret_cast<const u32x4*>(p1);
  return __builtin_bit_cast(v16bf, s);
}

static __device__ __forceinline__ v16bf ones_bf16() {
  Pack32 s;
  s.a = (u32x4)(0x3F803F80u);   // bf16 1.0 pair, splat
  s.b = (u32x4)(0x3F803F80u);
  return __builtin_bit_cast(v16bf, s);
}

static __device__ __forceinline__ unsigned short f2bf(float f) {
  unsigned int u = __float_as_uint(f);
  unsigned int r = u + 0x7fffu + ((u >> 16) & 1u);   // RNE
  return (unsigned short)(r >> 16);
}

static __device__ __forceinline__ v8f wmma_bf16(v16bf a, v16bf b, v8f c) {
  return __builtin_amdgcn_wmma_f32_16x16x32_bf16(false, a, false, b,
                                                 (short)0, c, false, false);
}

// ---------------- constants ----------------
#define DIMC   768
#define NTOK   2304      // 48*48
#define NB     2
#define NHEAD  12
#define HD     64
#define MROWS  (NB*NTOK) // 4608

// ---------------- cast f32 -> bf16 ----------------
__global__ void cast_bf16_k(const float* __restrict__ in,
                            unsigned short* __restrict__ out, int n) {
  int i = blockIdx.x * blockDim.x + threadIdx.x;
  if (i < n) out[i] = f2bf(in[i]);
}

// ---------------- wave-level 32x64 GEMM core ----------------
// K fully unrolled (K = DIMC = 768 -> 24 straight-line steps): every fragment
// is a distinct SSA value so the scheduler hoists loads freely with no
// accumulator copies or buffer rotation.
// acc[t] : rows 0-15 col tile t ; acc[4+t] : rows 16-31 col tile t
static __device__ __forceinline__ void gemm_tile32x64(
    const unsigned short* __restrict__ A, int lda,
    const unsigned short* __restrict__ W, int ldw,
    v8f acc[8]) {
  int lane = threadIdx.x & 31;
  int li = lane & 15, hi = lane >> 4;
  const unsigned short* ar0 = A + (size_t)li * lda;
  const unsigned short* ar1 = A + (size_t)(16 + li) * lda;
#pragma unroll
  for (int k = 0; k < DIMC; k += 32) {
    v16bf a0 = load_v16(ar0 + k + hi * 8, ar0 + k + 16 + hi * 8);
    v16bf a1 = load_v16(ar1 + k + hi * 8, ar1 + k + 16 + hi * 8);
    v16bf b[4];
#pragma unroll
    for (int t = 0; t < 4; ++t) {
      const unsigned short* wr = W + (size_t)(t * 16 + li) * ldw + k + hi * 8;
      b[t] = load_v16(wr, wr + 16);
    }
#pragma unroll
    for (int t = 0; t < 4; ++t) {
      acc[t]     = wmma_bf16(a0, b[t], acc[t]);
      acc[4 + t] = wmma_bf16(a1, b[t], acc[4 + t]);
    }
  }
}

// ---------------- q projection: out -> [B, h, N, hd] bf16 ----------------
__global__ __launch_bounds__(128) void gemm_qproj_k(
    const unsigned short* __restrict__ X, const unsigned short* __restrict__ W,
    const float* __restrict__ bias, unsigned short* __restrict__ Q) {
  const int NT = DIMC / 64;  // 12
  int wid = (blockIdx.x * blockDim.x + threadIdx.x) >> 5;
  if (wid >= (MROWS / 32) * NT) return;
  int mt = wid / NT, nt = wid % NT;
  int R = mt * 32, colb = nt * 64;
  v8f acc[8] = {};
  gemm_tile32x64(X + (size_t)R * DIMC, DIMC, W + (size_t)colb * DIMC, DIMC, acc);
  int lane = threadIdx.x & 31, li = lane & 15, hi = lane >> 4;
  int b = R / NTOK, n0 = R % NTOK;
#pragma unroll
  for (int t = 0; t < 4; ++t) {
    int col = colb + t * 16 + li;
    int h = col >> 6, d = col & 63;
    float bv = bias[col];
    unsigned short* dst = Q + ((size_t)(b * NHEAD + h) * NTOK + n0) * HD + d;
#pragma unroll
    for (int rb = 0; rb < 2; ++rb)
#pragma unroll
      for (int j = 0; j < 8; ++j)
        dst[(size_t)(rb * 16 + j + 8 * hi) * HD] = f2bf(acc[rb * 4 + t][j] + bv);
  }
}

// ---------------- kv projection: k -> [B,h,N,hd], v -> [B,h,hd,N] ----------
__global__ __launch_bounds__(128) void gemm_kvproj_k(
    const unsigned short* __restrict__ X, const unsigned short* __restrict__ W,
    const float* __restrict__ bias, unsigned short* __restrict__ Ko,
    unsigned short* __restrict__ VTo) {
  const int NT = (2 * DIMC) / 64;  // 24
  int wid = (blockIdx.x * blockDim.x + threadIdx.x) >> 5;
  if (wid >= (MROWS / 32) * NT) return;
  int mt = wid / NT, nt = wid % NT;
  int R = mt * 32, colb = nt * 64;
  v8f acc[8] = {};
  gemm_tile32x64(X + (size_t)R * DIMC, DIMC, W + (size_t)colb * DIMC, DIMC, acc);
  int lane = threadIdx.x & 31, li = lane & 15, hi = lane >> 4;
  int b = R / NTOK, n0 = R % NTOK;
#pragma unroll
  for (int t = 0; t < 4; ++t) {
    int col = colb + t * 16 + li;
    int two = col / DIMC, rem = col % DIMC;
    int h = rem >> 6, d = rem & 63;
    float bv = bias[col];
    if (two == 0) {
      unsigned short* dst = Ko + ((size_t)(b * NHEAD + h) * NTOK + n0) * HD + d;
#pragma unroll
      for (int rb = 0; rb < 2; ++rb)
#pragma unroll
        for (int j = 0; j < 8; ++j)
          dst[(size_t)(rb * 16 + j + 8 * hi) * HD] = f2bf(acc[rb * 4 + t][j] + bv);
    } else {
      unsigned short* dst = VTo + ((size_t)(b * NHEAD + h) * HD + d) * NTOK + n0;
#pragma unroll
      for (int rb = 0; rb < 2; ++rb)
#pragma unroll
        for (int j = 0; j < 8; ++j)
          dst[rb * 16 + j + 8 * hi] = f2bf(acc[rb * 4 + t][j] + bv);
    }
  }
}

// ---------------- flash attention: per wave 32 q-rows ----------------
__global__ __launch_bounds__(128) void attn_k(
    const unsigned short* __restrict__ Q, const unsigned short* __restrict__ Kb,
    const unsigned short* __restrict__ Vt, unsigned short* __restrict__ AO) {
  __shared__ unsigned short plds[4][32 * 32];
  const int QB = NTOK / 32;  // 72
  int wid = (blockIdx.x * blockDim.x + threadIdx.x) >> 5;
  if (wid >= NB * NHEAD * QB) return;
  int pair = wid / QB, qb = wid % QB;
  int lane = threadIdx.x & 31, li = lane & 15, hi = lane >> 4;
  unsigned short* pl = plds[(threadIdx.x >> 5) & 3];

  const unsigned short* qp = Q + ((size_t)pair * NTOK + qb * 32) * HD;
  const unsigned short* kp = Kb + (size_t)pair * NTOK * HD;
  const unsigned short* vp = Vt + (size_t)pair * HD * NTOK;

  const unsigned short* qr0 = qp + (size_t)li * HD;
  const unsigned short* qr1 = qp + (size_t)(16 + li) * HD;
  v16bf q0a = load_v16(qr0 + hi * 8, qr0 + 16 + hi * 8);        // rows 0-15, d 0..31
  v16bf q1a = load_v16(qr0 + 32 + hi * 8, qr0 + 48 + hi * 8);   // rows 0-15, d 32..63
  v16bf q0b = load_v16(qr1 + hi * 8, qr1 + 16 + hi * 8);        // rows 16-31
  v16bf q1b = load_v16(qr1 + 32 + hi * 8, qr1 + 48 + hi * 8);

  const v16bf onesf = ones_bf16();
  v8f acc[8] = {};
  v8f accL0 = {}, accL1 = {};   // row-sum accumulators (P @ ones)
  float mr0[8], mr1[8];
#pragma unroll
  for (int j = 0; j < 8; ++j) { mr0[j] = -3.0e38f; mr1[j] = -3.0e38f; }
  const float scale = 0.125f;  // 64^-0.5

  auto loadK = [&](int kb, v16bf (&kf)[4]) {
    const unsigned short* kr0 = kp + (size_t)(kb + li) * HD;
    const unsigned short* kr1 = kp + (size_t)(kb + 16 + li) * HD;
    kf[0] = load_v16(kr0 + hi * 8, kr0 + 16 + hi * 8);         // k rows kb..+15, d 0..31
    kf[1] = load_v16(kr0 + 32 + hi * 8, kr0 + 48 + hi * 8);   // d 32..63
    kf[2] = load_v16(kr1 + hi * 8, kr1 + 16 + hi * 8);         // k rows kb+16..+31
    kf[3] = load_v16(kr1 + 32 + hi * 8, kr1 + 48 + hi * 8);
  };
  // per 16x32 S block: running max, alpha, write exp(P) to LDS
  auto softmax32 = [&](v8f sA, v8f sB, float* mr, float* al, int rbase) {
#pragma unroll
    for (int j = 0; j < 8; ++j) {
      float a = sA[j] * scale, b2 = sB[j] * scale;
      float t = fmaxf(a, b2);
#pragma unroll
      for (int off = 1; off < 16; off <<= 1) t = fmaxf(t, __shfl_xor(t, off));
      float mnew = fmaxf(mr[j], t);
      al[j] = __expf(mr[j] - mnew);
      mr[j] = mnew;
      int r = rbase + j + 8 * hi;
      pl[r * 32 + li]      = f2bf(__expf(a - mnew));
      pl[r * 32 + 16 + li] = f2bf(__expf(b2 - mnew));
    }
  };

  // one 32-k-row step; consumes kf (resident); PF: prefetch kb+32 into kfn
  auto attnStep = [&](int kb, v16bf (&kf)[4], v16bf (&kfn)[4], bool pf) {
    v16bf vf[4];
#pragma unroll
    for (int dt = 0; dt < 4; ++dt) {
      const unsigned short* vr = vp + (size_t)(dt * 16 + li) * NTOK + kb + hi * 8;
      vf[dt] = load_v16(vr, vr + 16);
    }
    if (pf) loadK(kb + 32, kfn);

    v8f s0 = {}; s0 = wmma_bf16(q0a, kf[0], s0); s0 = wmma_bf16(q1a, kf[1], s0);
    v8f s1 = {}; s1 = wmma_bf16(q0a, kf[2], s1); s1 = wmma_bf16(q1a, kf[3], s1);
    v8f s2 = {}; s2 = wmma_bf16(q0b, kf[0], s2); s2 = wmma_bf16(q1b, kf[1], s2);
    v8f s3 = {}; s3 = wmma_bf16(q0b, kf[2], s3); s3 = wmma_bf16(q1b, kf[3], s3);

    float al0[8], al1[8];
    softmax32(s0, s1, mr0, al0, 0);
    softmax32(s2, s3, mr1, al1, 16);

    asm volatile("" ::: "memory");  // keep P stores before A-frag reads (DS in-order)
    const unsigned short* pr0 = pl + li * 32;
    const unsigned short* pr1 = pl + (size_t)(16 + li) * 32;
    v16bf pa0 = load_v16(pr0 + hi * 8, pr0 + 16 + hi * 8);
    v16bf pa1 = load_v16(pr1 + hi * 8, pr1 + 16 + hi * 8);

#pragma unroll
    for (int j = 0; j < 8; ++j) {
      accL0[j] *= al0[j];
      accL1[j] *= al1[j];
#pragma unroll
      for (int t4 = 0; t4 < 4; ++t4) {
        acc[t4][j]     *= al0[j];
        acc[4 + t4][j] *= al1[j];
      }
    }
#pragma unroll
    for (int dt = 0; dt < 4; ++dt) {
      acc[dt]     = wmma_bf16(pa0, vf[dt], acc[dt]);
      acc[4 + dt] = wmma_bf16(pa1, vf[dt], acc[4 + dt]);
    }
    accL0 = wmma_bf16(pa0, onesf, accL0);   // row sums via matrix pipe
    accL1 = wmma_bf16(pa1, onesf, accL1);
  };

  v16bf kfA[4], kfB[4];
  loadK(0, kfA);
  int kb = 0;
  // main loop: prefetch is unconditional (no masked paths in hot loop)
  for (; kb < NTOK - 64; kb += 64) {
    attnStep(kb,      kfA, kfB, true);
    attnStep(kb + 32, kfB, kfA, true);
  }
  // peeled tail
  attnStep(kb,      kfA, kfB, true);
  attnStep(kb + 32, kfB, kfA, false);

  int b = pair / NHEAD, h = pair % NHEAD;
#pragma unroll
  for (int dt = 0; dt < 4; ++dt) {
    int col = h * 64 + dt * 16 + li;
    unsigned short* dst = AO + ((size_t)b * NTOK + qb * 32) * DIMC + col;
#pragma unroll
    for (int j = 0; j < 8; ++j) {
      int r0 = j + 8 * hi;
      dst[(size_t)r0 * DIMC]        = f2bf(acc[dt][j] / accL0[j]);
      dst[(size_t)(16 + r0) * DIMC] = f2bf(acc[4 + dt][j] / accL1[j]);
    }
  }
}

// ---------------- output projection: f32 out ----------------
__global__ __launch_bounds__(128) void gemm_oproj_k(
    const unsigned short* __restrict__ X, const unsigned short* __restrict__ W,
    const float* __restrict__ bias, float* __restrict__ O) {
  const int NT = DIMC / 64;
  int wid = (blockIdx.x * blockDim.x + threadIdx.x) >> 5;
  if (wid >= (MROWS / 32) * NT) return;
  int mt = wid / NT, nt = wid % NT;
  int R = mt * 32, colb = nt * 64;
  v8f acc[8] = {};
  gemm_tile32x64(X + (size_t)R * DIMC, DIMC, W + (size_t)colb * DIMC, DIMC, acc);
  int lane = threadIdx.x & 31, li = lane & 15, hi = lane >> 4;
#pragma unroll
  for (int t = 0; t < 4; ++t) {
    int col = colb + t * 16 + li;
    float bv = bias[col];
#pragma unroll
    for (int rb = 0; rb < 2; ++rb)
#pragma unroll
      for (int j = 0; j < 8; ++j)
        O[(size_t)(R + rb * 16 + j + 8 * hi) * DIMC + col] = acc[rb * 4 + t][j] + bv;
  }
}

// ---------------- channel mean / max per token ----------------
__global__ void rowstat_k(const float* __restrict__ P, float* __restrict__ avg,
                          float* __restrict__ mx) {
  __shared__ float ssum[256], smax[256];
  int row = blockIdx.x;
  const float* p = P + (size_t)row * DIMC;
  int t = threadIdx.x;
  float s = 0.f, m = -3.0e38f;
  for (int c = t; c < DIMC; c += 256) { float v = p[c]; s += v; m = fmaxf(m, v); }
  ssum[t] = s; smax[t] = m; __syncthreads();
  for (int o = 128; o > 0; o >>= 1) {
    if (t < o) { ssum[t] += ssum[t + o]; smax[t] = fmaxf(smax[t], smax[t + o]); }
    __syncthreads();
  }
  if (t == 0) { avg[row] = ssum[0] / (float)DIMC; mx[row] = smax[0]; }
}

// ---------------- 7x7 2-in 1-out conv + sigmoid ----------------
__global__ void conv_sig_k(const float* __restrict__ avg, const float* __restrict__ mx,
                           const float* __restrict__ w, float* __restrict__ sig) {
  int idx = blockIdx.x * blockDim.x + threadIdx.x;
  if (idx >= NB * NTOK) return;
  int b = idx / NTOK, n = idx % NTOK;
  int y = n / 48, x = n % 48;
  float a = 0.f;
  for (int ci = 0; ci < 2; ++ci) {
    const float* pin = (ci == 0 ? avg : mx) + (size_t)b * NTOK;
    for (int ky = 0; ky < 7; ++ky) {
      int yy = y + ky - 3; if (yy < 0 || yy >= 48) continue;
      for (int kx = 0; kx < 7; ++kx) {
        int xx = x + kx - 3; if (xx < 0 || xx >= 48) continue;
        a += pin[yy * 48 + xx] * w[(ci * 7 + ky) * 7 + kx];
      }
    }
  }
  sig[idx] = 1.f / (1.f + __expf(-a));
}

// ---------------- broadcast multiply ----------------
__global__ void finalmul_k(const float* __restrict__ P, const float* __restrict__ sig,
                           float* __restrict__ out, int n) {
  int i = blockIdx.x * blockDim.x + threadIdx.x;
  if (i < n) out[i] = P[i] * sig[i / DIMC];
}

// ---------------- launch ----------------
extern "C" void kernel_launch(void* const* d_in, const int* in_sizes, int n_in,
                              void* d_out, int out_size, void* d_ws, size_t ws_size,
                              hipStream_t stream) {
  const float* q_input = (const float*)d_in[0];
  const float* kv_input = (const float*)d_in[1];
  const float* q_w  = (const float*)d_in[2];
  const float* q_b  = (const float*)d_in[3];
  const float* kv_w = (const float*)d_in[4];
  const float* kv_b = (const float*)d_in[5];
  const float* p_w  = (const float*)d_in[6];
  const float* p_b  = (const float*)d_in[7];
  const float* sa_w = (const float*)d_in[8];
  float* out = (float*)d_out;

  // workspace carve-out (256B aligned)
  uintptr_t base = (uintptr_t)d_ws;
  auto carve = [&](size_t bytes) {
    uintptr_t p = (base + 255) & ~(uintptr_t)255;
    base = p + bytes;
    return (void*)p;
  };
  const size_t MC = (size_t)MROWS * DIMC;
  unsigned short* Xq  = (unsigned short*)carve(MC * 2);
  unsigned short* Xkv = (unsigned short*)carve(MC * 2);
  unsigned short* Wq  = (unsigned short*)carve((size_t)DIMC * DIMC * 2);
  unsigned short* Wkv = (unsigned short*)carve((size_t)2 * DIMC * DIMC * 2);
  unsigned short* Wp  = (unsigned short*)carve((size_t)DIMC * DIMC * 2);
  unsigned short* Qp  = (unsigned short*)carve(MC * 2);   // [B,h,N,hd]
  unsigned short* Kp  = (unsigned short*)carve(MC * 2);   // [B,h,N,hd]
  unsigned short* VTp = (unsigned short*)carve(MC * 2);   // [B,h,hd,N]
  unsigned short* AO  = (unsigned short*)carve(MC * 2);   // [B,N,C]
  float* Pout = (float*)carve(MC * 4);
  float* Avg  = (float*)carve((size_t)MROWS * 4);
  float* Mx   = (float*)carve((size_t)MROWS * 4);
  float* Sig  = (float*)carve((size_t)MROWS * 4);

  // casts
  auto cast = [&](const float* in, unsigned short* o, int n) {
    cast_bf16_k<<<(n + 255) / 256, 256, 0, stream>>>(in, o, n);
  };
  cast(q_input, Xq, (int)MC);
  cast(kv_input, Xkv, (int)MC);
  cast(q_w, Wq, DIMC * DIMC);
  cast(kv_w, Wkv, 2 * DIMC * DIMC);
  cast(p_w, Wp, DIMC * DIMC);

  // projections
  {
    int waves = (MROWS / 32) * (DIMC / 64);          // 1728
    gemm_qproj_k<<<(waves * 32 + 127) / 128, 128, 0, stream>>>(Xq, Wq, q_b, Qp);
  }
  {
    int waves = (MROWS / 32) * (2 * DIMC / 64);      // 3456
    gemm_kvproj_k<<<(waves * 32 + 127) / 128, 128, 0, stream>>>(Xkv, Wkv, kv_b, Kp, VTp);
  }
  // attention
  {
    int waves = NB * NHEAD * (NTOK / 32);            // 1728
    attn_k<<<(waves * 32 + 127) / 128, 128, 0, stream>>>(Qp, Kp, VTp, AO);
  }
  // out projection
  {
    int waves = (MROWS / 32) * (DIMC / 64);
    gemm_oproj_k<<<(waves * 32 + 127) / 128, 128, 0, stream>>>(AO, Wp, p_b, Pout);
  }
  // spatial attention branch
  rowstat_k<<<MROWS, 256, 0, stream>>>(Pout, Avg, Mx);
  conv_sig_k<<<(NB * NTOK + 255) / 256, 256, 0, stream>>>(Avg, Mx, sa_w, Sig);
  finalmul_k<<<((int)MC + 255) / 256, 256, 0, stream>>>(Pout, Sig, out, (int)MC);
}
